// RelativeHead_59030030516232
// MI455X (gfx1250) — matrix operand order repaired and checked by reading, not compiled, not run on hardware
//
#include <hip/hip_runtime.h>
#include <hip/hip_bf16.h>
#include <math.h>

#define BB 8
#define TT 2048
#define CC 1024
#define HH 64

typedef __attribute__((ext_vector_type(16))) __bf16 v16bf;
typedef __attribute__((ext_vector_type(8)))  __bf16 v8bf;   // one 16B load chunk
typedef __attribute__((ext_vector_type(8)))  float  v8f;

union AF { v16bf v; v8bf h[2]; __bf16 e[16]; };
union CF { v8f  v; float f[8]; };

__device__ __forceinline__ v8f wmma_bf16(v16bf a, v16bf b, v8f c) {
  return __builtin_amdgcn_wmma_f32_16x16x32_bf16(false, a, false, b, (short)0, c,
                                                 false, false);
}
__device__ __forceinline__ v8bf ldg8(const __bf16* p) { return *(const v8bf*)p; }

// ---------------- E -> bf16 ----------------
__global__ void __launch_bounds__(256) ecvt_kernel(const float* __restrict__ E,
                                                   __bf16* __restrict__ Eb) {
  int i = blockIdx.x * 256 + threadIdx.x;
  if (i < TT * HH) Eb[i] = (__bf16)E[i];
}

// ---------------- Wq -> WqT bf16 ([H][C]) ----------------
__global__ void __launch_bounds__(256) wcvt_kernel(const float* __restrict__ Wq,
                                                   __bf16* __restrict__ WqT) {
  int i = blockIdx.x * 256 + threadIdx.x;  // i = h*CC + c
  if (i < CC * HH) {
    int h = i / CC, c = i % CC;
    WqT[i] = (__bf16)Wq[(size_t)c * HH + h];
  }
}

// ---------------- q = x @ Wq (bf16 WMMA, f32 acc); also emit qT ----------------
__global__ void __launch_bounds__(32)
qproj_kernel(const float* __restrict__ x, const __bf16* __restrict__ WqT,
             __bf16* __restrict__ qb, __bf16* __restrict__ qT) {
  const int lane = threadIdx.x & 31;
  const int half = lane >> 4;
  const int lidx = lane & 15;
  const int mt = blockIdx.x >> 2;  // B*T/16 tiles
  const int nt = blockIdx.x & 3;   // H/16 tiles

  const float*  xrow = x + (size_t)(mt * 16 + lidx) * CC;        // A row (lane)
  const __bf16* wrow = WqT + (size_t)(nt * 16 + lidx) * CC;      // B col -> WqT row

  CF c;
#pragma unroll
  for (int r = 0; r < 8; ++r) c.f[r] = 0.0f;

  for (int k0 = 0; k0 < CC; k0 += 32) {
    AF a, b;
    // A: two contiguous 8-float chunks at K = {half*8, 16+half*8}
    const float* c0 = xrow + k0 + (half << 3);
    const float* c1 = c0 + 16;
    float4 u0 = *(const float4*)c0, u1 = *(const float4*)(c0 + 4);
    float4 u2 = *(const float4*)c1, u3 = *(const float4*)(c1 + 4);
    a.e[0] = (__bf16)u0.x; a.e[1] = (__bf16)u0.y; a.e[2] = (__bf16)u0.z; a.e[3] = (__bf16)u0.w;
    a.e[4] = (__bf16)u1.x; a.e[5] = (__bf16)u1.y; a.e[6] = (__bf16)u1.z; a.e[7] = (__bf16)u1.w;
    a.e[8]  = (__bf16)u2.x; a.e[9]  = (__bf16)u2.y; a.e[10] = (__bf16)u2.z; a.e[11] = (__bf16)u2.w;
    a.e[12] = (__bf16)u3.x; a.e[13] = (__bf16)u3.y; a.e[14] = (__bf16)u3.z; a.e[15] = (__bf16)u3.w;
    // B: two contiguous 8-bf16 chunks at K = {half*8, 16+half*8}
    b.h[0] = ldg8(wrow + k0 + (half << 3));
    b.h[1] = ldg8(wrow + k0 + 16 + (half << 3));
    c.v = wmma_bf16(a.v, b.v, c.v);
  }
#pragma unroll
  for (int r = 0; r < 8; ++r) {
    size_t row = (size_t)mt * 16 + r + (half << 3);  // global B*T row
    __bf16 qv = (__bf16)c.f[r];
    qb[row * HH + nt * 16 + lidx] = qv;
    size_t bidx = row / TT, t = row % TT;
    qT[((size_t)bidx * HH + nt * 16 + lidx) * TT + t] = qv;  // transposed copy
  }
}

// ---------------- flash attention with skewed relative bias ----------------
__global__ void __launch_bounds__(32)
attn_kernel(const __bf16* __restrict__ qb, const __bf16* __restrict__ qT,
            const __bf16* __restrict__ Eb, float* __restrict__ out) {
  __shared__ float  lds_w[2][16][32];  // bias W tiles (double-buffered per key tile)
  __shared__ __bf16 lds_p[16][32];     // probs staging (C-frag -> A-frag re-layout)

  const int lane = threadIdx.x & 31;
  const int half = lane >> 4;
  const int lidx = lane & 15;
  const int qt = blockIdx.x & 127;  // T/16 = 128 query tiles
  const int b  = blockIdx.x >> 7;
  const int i0 = qt * 16;
  const __bf16* qbase  = qb + (size_t)b * TT * HH;
  const __bf16* qTbase = qT + (size_t)b * HH * TT;

  // Query A-fragments: row i0+lidx; chunks at h = {half*8, 16+half*8} (+32)
  AF aq0, aq1;
  {
    const __bf16* qrow = qbase + (size_t)(i0 + lidx) * HH;
    aq0.h[0] = ldg8(qrow + (half << 3));
    aq0.h[1] = ldg8(qrow + 16 + (half << 3));
    aq1.h[0] = ldg8(qrow + 32 + (half << 3));
    aq1.h[1] = ldg8(qrow + 48 + (half << 3));
  }

  CF o[4];
#pragma unroll
  for (int n = 0; n < 4; ++n)
#pragma unroll
    for (int r = 0; r < 8; ++r) o[n].f[r] = 0.0f;
  float m[8], lsum[8];
#pragma unroll
  for (int r = 0; r < 8; ++r) { m[r] = -INFINITY; lsum[r] = 0.0f; }

  const float scale = 0.125f;       // H^-0.5
  const int npair = (qt >> 1) + 1;  // 32-key pairs covering [0, i0+16)

  for (int pr = 0; pr < npair; ++pr) {
    const int ka = pr * 32;
    float st[2][8];
    float mx[8];
#pragma unroll
    for (int r = 0; r < 8; ++r) mx[r] = -INFINITY;

    for (int s = 0; s < 2; ++s) {
      const int j0 = ka + s * 16;
      // ---- S = q qT : key B-frag = contiguous q row of key j0+lidx ----
      AF bq0, bq1;
      {
        const __bf16* krow = qbase + (size_t)(j0 + lidx) * HH;
        bq0.h[0] = ldg8(krow + (half << 3));
        bq0.h[1] = ldg8(krow + 16 + (half << 3));
        bq1.h[0] = ldg8(krow + 32 + (half << 3));
        bq1.h[1] = ldg8(krow + 48 + (half << 3));
      }
      CF sc;
#pragma unroll
      for (int r = 0; r < 8; ++r) sc.f[r] = 0.0f;
      sc.v = wmma_bf16(aq0.v, bq0.v, sc.v);
      sc.v = wmma_bf16(aq1.v, bq1.v, sc.v);

      // ---- bias: W[ii,c] = q[i0+ii] . E[d0-15+c], c in [0,32) ----
      const int d0 = TT - 1 - i0 + j0;
#pragma unroll
      for (int wt = 0; wt < 2; ++wt) {
        int erow = d0 - 15 + wt * 16 + lidx;  // this lane's B column
        erow = erow < 0 ? 0 : (erow > TT - 1 ? TT - 1 : erow);
        const __bf16* er = Eb + (size_t)erow * HH;
        AF be0, be1;
        be0.h[0] = ldg8(er + (half << 3));
        be0.h[1] = ldg8(er + 16 + (half << 3));
        be1.h[0] = ldg8(er + 32 + (half << 3));
        be1.h[1] = ldg8(er + 48 + (half << 3));
        CF w;
#pragma unroll
        for (int r = 0; r < 8; ++r) w.f[r] = 0.0f;
        w.v = wmma_bf16(aq0.v, be0.v, w.v);
        w.v = wmma_bf16(aq1.v, be1.v, w.v);
#pragma unroll
        for (int r = 0; r < 8; ++r)
          lds_w[s][r + (half << 3)][wt * 16 + lidx] = w.f[r];
      }
      __syncthreads();

      // ---- skew gather + scale + causal mask ----
#pragma unroll
      for (int r = 0; r < 8; ++r) {
        int ii = r + (half << 3);
        int jj = lidx;
        float bias = lds_w[s][ii][15 + jj - ii];  // always in [0,30]
        float v = (sc.f[r] + bias) * scale;
        if (j0 + jj > i0 + ii) v = -INFINITY;
        st[s][r] = v;
        mx[r] = fmaxf(mx[r], v);
      }
    }

    // ---- online softmax over the 32 keys ----
    float pv[2][8], resc[8];
#pragma unroll
    for (int r = 0; r < 8; ++r) {
      float v = mx[r];
      v = fmaxf(v, __shfl_xor(v, 1, 32));
      v = fmaxf(v, __shfl_xor(v, 2, 32));
      v = fmaxf(v, __shfl_xor(v, 4, 32));
      v = fmaxf(v, __shfl_xor(v, 8, 32));  // row max (stays within 16-lane half)
      float mnew = fmaxf(m[r], v);
      resc[r] = __expf(m[r] - mnew);
      m[r] = mnew;
      float p0 = __expf(st[0][r] - mnew);
      float p1 = __expf(st[1][r] - mnew);
      pv[0][r] = p0;
      pv[1][r] = p1;
      float rs = p0 + p1;
      rs += __shfl_xor(rs, 1, 32);
      rs += __shfl_xor(rs, 2, 32);
      rs += __shfl_xor(rs, 4, 32);
      rs += __shfl_xor(rs, 8, 32);
      lsum[r] = lsum[r] * resc[r] + rs;
    }
#pragma unroll
    for (int n = 0; n < 4; ++n)
#pragma unroll
      for (int r = 0; r < 8; ++r) o[n].f[r] *= resc[r];

    // ---- P: C-layout -> A-fragment via LDS (bf16) ----
#pragma unroll
    for (int s2 = 0; s2 < 2; ++s2)
#pragma unroll
      for (int r = 0; r < 8; ++r)
        lds_p[r + (half << 3)][s2 * 16 + lidx] = (__bf16)pv[s2][r];
    __syncthreads();
    AF pa;
    pa.h[0] = *(const v8bf*)&lds_p[lidx][half << 3];        // K = half*8..+7
    pa.h[1] = *(const v8bf*)&lds_p[lidx][16 + (half << 3)]; // K = 16+half*8..+7

    // ---- O += P @ V : V B-frag from qT (contiguous over keys) ----
#pragma unroll
    for (int n = 0; n < 4; ++n) {
      AF bv;
      const __bf16* vrow = qTbase + (size_t)(n * 16 + lidx) * TT + ka;
      bv.h[0] = ldg8(vrow + (half << 3));
      bv.h[1] = ldg8(vrow + 16 + (half << 3));
      o[n].v = wmma_bf16(pa.v, bv.v, o[n].v);
    }
    __syncthreads();  // before LDS reuse next pair
  }

  // ---- normalize + store ----
#pragma unroll
  for (int r = 0; r < 8; ++r) {
    float inv = 1.0f / lsum[r];
    size_t row = (size_t)b * TT + i0 + r + (half << 3);
#pragma unroll
    for (int n = 0; n < 4; ++n)
      out[row * HH + n * 16 + lidx] = o[n].f[r] * inv;
  }
}

extern "C" void kernel_launch(void* const* d_in, const int* in_sizes, int n_in,
                              void* d_out, int out_size, void* d_ws, size_t ws_size,
                              hipStream_t stream) {
  (void)in_sizes; (void)n_in; (void)out_size; (void)ws_size;
  const float* x  = (const float*)d_in[0];
  const float* Wq = (const float*)d_in[1];
  const float* E  = (const float*)d_in[2];
  float* out = (float*)d_out;

  __bf16* qb  = (__bf16*)d_ws;                      // [B*T, H] bf16 : 2 MB
  __bf16* qT  = qb + (size_t)BB * TT * HH;          // [B, H, T] bf16 : 2 MB
  __bf16* Eb  = qT + (size_t)BB * TT * HH;          // [T, H]   bf16 : 256 KB
  __bf16* WqT = Eb + (size_t)TT * HH;               // [H, C]   bf16 : 128 KB

  ecvt_kernel<<<(TT * HH + 255) / 256, 256, 0, stream>>>(E, Eb);
  wcvt_kernel<<<(CC * HH + 255) / 256, 256, 0, stream>>>(Wq, WqT);
  qproj_kernel<<<(BB * TT / 16) * (HH / 16), 32, 0, stream>>>(x, WqT, qb, qT);
  attn_kernel<<<BB * (TT / 16), 32, 0, stream>>>(qb, qT, Eb, out);
}
